// TweetyBERT_8375186227933
// MI455X (gfx1250) — matrix operand order repaired and checked by reading, not compiled, not run on hardware
//
#include <hip/hip_runtime.h>
#include <hip/hip_bf16.h>

// ---------------------------------------------------------------------------
// TweetyBERT forward on gfx1250 (MI455X).
// Compute-bound (~0.73 TFLOP vs ~hundreds of MB of traffic at 23.3 TB/s), so
// all dense math (convs 2-4 as per-tap implicit GEMM, all transformer GEMMs,
// fused attention) runs through v_wmma_f32_16x16x32_bf16 with f32 accumulate.
// GEMM staging uses gfx1250 async global->LDS (ASYNCcnt) with double buffering.
// ---------------------------------------------------------------------------

typedef __bf16 bf16;
typedef __attribute__((ext_vector_type(16))) __bf16 v16bf;
typedef __attribute__((ext_vector_type(8)))  __bf16 v8bf;
typedef __attribute__((ext_vector_type(8)))  float  v8f;

union V16 { v16bf v; v8bf h[2]; };

static __device__ __forceinline__ v16bf bzero16() {
  V16 u;
#pragma unroll
  for (int i = 0; i < 16; ++i) u.v[i] = (__bf16)0.f;
  return u.v;
}
static __device__ __forceinline__ v8bf bzero8() {
  v8bf v;
#pragma unroll
  for (int i = 0; i < 8; ++i) v[i] = (__bf16)0.f;
  return v;
}
static __device__ __forceinline__ v8f fzero8() {
  v8f v;
#pragma unroll
  for (int i = 0; i < 8; ++i) v[i] = 0.f;
  return v;
}

static __device__ __forceinline__ v8f wmma_bf16(v16bf a, v16bf b, v8f c) {
  // D = A(16x32) * B(32x16) + C, f32 accumulate
  return __builtin_amdgcn_wmma_f32_16x16x32_bf16(false, a, false, b,
                                                 (short)0, c, false, false);
}

static __device__ __forceinline__ float gelu(float x) {
  return 0.5f * x * (1.0f + erff(x * 0.70710678118654752f));
}

// ---------------------------------------------------------------------------
// gfx1250 async global->LDS copy (16 bytes / lane). The builtin's prototype
// (from the toolchain diagnostic) is:
//   void (v4i32 addrspace(1)* src, v4i32 addrspace(3)* dst, Ii offset, Ii cpol)
// Pointers are built from integers: global generic == AS1 address; the low 32
// bits of a generic LDS address are the LDS byte offset per ISA 10.2.
// Falls back to a synchronous vector copy if the builtin is not declared.
// ---------------------------------------------------------------------------
#if defined(__AMDGCN__) && __has_builtin(__builtin_amdgcn_global_load_async_to_lds_b128)
#define TB_ASYNC 1
#else
#define TB_ASYNC 0
#endif

typedef int tb_v4i __attribute__((vector_size(16)));

static __device__ __forceinline__ void lds_copy16(bf16* lds, const bf16* g) {
#if TB_ASYNC
  __builtin_amdgcn_global_load_async_to_lds_b128(
      (__attribute__((address_space(1))) tb_v4i*)(uintptr_t)g,
      (__attribute__((address_space(3))) tb_v4i*)(unsigned)(uintptr_t)lds,
      0, 0);
#else
  *(v8bf*)lds = *(const v8bf*)g;
#endif
}

static __device__ __forceinline__ void wait_async_all() {
#if TB_ASYNC
#if __has_builtin(__builtin_amdgcn_s_wait_asynccnt)
  __builtin_amdgcn_s_wait_asynccnt(0);
#else
  asm volatile("s_wait_asynccnt 0x0" ::: "memory");
#endif
#endif
}

// ---------------------------------------------------------------------------
// Elementwise / repack helpers
// ---------------------------------------------------------------------------

// fp32 W[K][N] -> bf16 WT[N][K]  (so WMMA B-fragments are contiguous per lane)
__global__ void k_wt(const float* __restrict__ W, bf16* __restrict__ WT,
                     int K, int N) {
  size_t e = (size_t)blockIdx.x * 256 + threadIdx.x;
  if (e >= (size_t)K * N) return;
  int n = (int)(e / K);
  int k = (int)(e % K);
  WT[e] = (bf16)W[(size_t)k * N + n];
}

__global__ void k_cvt(const float* __restrict__ in, bf16* __restrict__ out,
                      size_t n) {
  size_t e = (size_t)blockIdx.x * 256 + threadIdx.x;
  if (e < n) out[e] = (bf16)in[e];
}

// Pack conv weights W[Cout][Cin][5][5] (fp32) into exact A-fragment order:
// pack[(((tap*KS+ks)*4+mt)*32+lane)*16+j], A 16x32 bf16, contiguous k order k = 16*(lane/16) + j (B matches).
__global__ void k_pack_conv(const float* __restrict__ W, bf16* __restrict__ pack,
                            int CIN) {
  int KS = CIN / 32;
  int total = 25 * KS * 4 * 512;
  int e = blockIdx.x * 256 + threadIdx.x;
  if (e >= total) return;
  int j    = e & 15;
  int lane = (e >> 4) & 31;
  int mt   = (e >> 9) & 3;
  int rest = e >> 11;
  int ks   = rest % KS;
  int tap  = rest / KS;
  int K_   = j + ((lane < 16) ? 0 : 16);
  int ci   = ks * 32 + K_;
  int co   = mt * 16 + (lane & 15);
  int dh   = tap / 5, dt = tap % 5;
  pack[e] = (bf16)W[((size_t)(co * CIN + ci) * 5 + dh) * 5 + dt];
}

// X4[b][h][t][c] (channels-last, 12x64) -> tokens [b*T+t][c*12+h] bf16
__global__ void k_tok(const bf16* __restrict__ X4, bf16* __restrict__ tok,
                      int T) {
  size_t e = (size_t)blockIdx.x * 256 + threadIdx.x;
  size_t total = (size_t)16 * T * 768;
  if (e >= total) return;
  int f = (int)(e % 768);
  int t = (int)((e / 768) % T);
  int b = (int)(e / ((size_t)768 * T));
  int c = f / 12, h = f % 12;
  tok[e] = X4[(((size_t)b * 12 + h) * T + t) * 64 + c];
}

// v_bf [B,T,768] -> VT[(b*12+h)*64+dh][512] bf16, zero-padded past T
__global__ void k_vt(const bf16* __restrict__ vb, bf16* __restrict__ VT, int T) {
  size_t e = (size_t)blockIdx.x * 256 + threadIdx.x;
  const size_t total = (size_t)16 * 12 * 64 * 512;
  if (e >= total) return;
  int t  = (int)(e & 511);
  int dh = (int)((e >> 9) & 63);
  int bh = (int)(e >> 15);
  int h = bh % 12, b = bh / 12;
  VT[e] = (t < T) ? vb[((size_t)(b * T + t)) * 768 + h * 64 + dh] : (bf16)0.f;
}

// ---------------------------------------------------------------------------
// Conv1 (Cin=1): direct fp32, fused bias + GELU + freq-maxpool -> bf16 CL
// ---------------------------------------------------------------------------
__global__ __launch_bounds__(128) void k_conv1(
    const float* __restrict__ spec, const float* __restrict__ w,
    const float* __restrict__ cb, bf16* __restrict__ X1, int T) {
  __shared__ float sw[800];
  __shared__ float sb[32];
  for (int i = threadIdx.x; i < 800; i += 128) sw[i] = w[i];
  if (threadIdx.x < 32) sb[threadIdx.x] = cb[threadIdx.x];
  __syncthreads();
  int t = blockIdx.x * 128 + threadIdx.x;
  int ho = blockIdx.y, b = blockIdx.z;
  if (t >= T) return;
  float accA[32], accB[32];
#pragma unroll
  for (int c = 0; c < 32; ++c) { accA[c] = sb[c]; accB[c] = sb[c]; }
  const float* sp = spec + (size_t)b * 192 * T;
  for (int rr = 0; rr < 6; ++rr) {
    int hin = 2 * ho - 2 + rr;
    float in[5];
#pragma unroll
    for (int dt = 0; dt < 5; ++dt) {
      int tt = t + dt - 2;
      in[dt] = (hin >= 0 && hin < 192 && tt >= 0 && tt < T)
                   ? sp[(size_t)hin * T + tt] : 0.f;
    }
#pragma unroll
    for (int dt = 0; dt < 5; ++dt) {
      float xv = in[dt];
      if (rr < 5) {
        int wi = rr * 5 + dt;
#pragma unroll
        for (int c = 0; c < 32; ++c) accA[c] = fmaf(sw[c * 25 + wi], xv, accA[c]);
      }
      if (rr > 0) {
        int wi = (rr - 1) * 5 + dt;
#pragma unroll
        for (int c = 0; c < 32; ++c) accB[c] = fmaf(sw[c * 25 + wi], xv, accB[c]);
      }
    }
  }
  bf16* out = X1 + (((size_t)b * 96 + ho) * T + t) * 32;
#pragma unroll
  for (int c = 0; c < 32; ++c)
    out[c] = (bf16)fmaxf(gelu(accA[c]), gelu(accB[c]));
}

// ---------------------------------------------------------------------------
// Convs 2-4: per-tap implicit GEMM with WMMA. 4 waves = 4 Cout tiles of 16.
// Two pre-pool rows accumulated, fused bias+GELU+maxpool epilogue.
// ---------------------------------------------------------------------------
template <int CIN, int HIN>
__global__ __launch_bounds__(128) void k_conv_wmma(
    const bf16* __restrict__ Xin, const bf16* __restrict__ Wpack,
    const float* __restrict__ cbias, bf16* __restrict__ Xout, int T) {
  const int lane = threadIdx.x & 31, wid = threadIdx.x >> 5;
  const int t0 = blockIdx.x * 16;
  const int ho = blockIdx.y;
  const int b  = blockIdx.z;
  const int HO = HIN / 2;
  const int KS = CIN / 32;
  v8f accA = fzero8(), accB = fzero8();
  const int tl  = t0 + (lane & 15);
  const int ci0 = (lane >> 4) * 16;
#pragma unroll
  for (int dh = 0; dh < 5; ++dh) {
    int hA = 2 * ho + dh - 2;
    int hB = hA + 1;
    bool okA = (hA >= 0) && (hA < HIN);
    bool okB = (hB >= 0) && (hB < HIN);
#pragma unroll
    for (int dt = 0; dt < 5; ++dt) {
      int t = tl + dt - 2;
      bool tok = (t >= 0) && (t < T);
#pragma unroll
      for (int ks = 0; ks < KS; ++ks) {
        v16bf va = *(const v16bf*)(Wpack +
            ((size_t)(((dh * 5 + dt) * KS + ks) * 4 + wid) * 32 + lane) * 16);
        v16bf vbA = (tok && okA)
            ? *(const v16bf*)(Xin + (((size_t)b * HIN + hA) * T + t) * CIN + ks * 32 + ci0)
            : bzero16();
        accA = wmma_bf16(va, vbA, accA);
        v16bf vbB = (tok && okB)
            ? *(const v16bf*)(Xin + (((size_t)b * HIN + hB) * T + t) * CIN + ks * 32 + ci0)
            : bzero16();
        accB = wmma_bf16(va, vbB, accB);
      }
    }
  }
#pragma unroll
  for (int r = 0; r < 8; ++r) {
    int co = wid * 16 + r + (lane >> 4) * 8;
    int t = t0 + (lane & 15);
    if (t < T) {
      float bv = cbias[co];
      float a = gelu(accA[r] + bv);
      float c = gelu(accB[r] + bv);
      Xout[(((size_t)b * HO + ho) * T + t) * 64 + co] = (bf16)fmaxf(a, c);
    }
  }
}

// ---------------------------------------------------------------------------
// Generic bf16 GEMM: C[M,N] = A[M,K] @ B[K,N] (+bias, epilogue variants)
// BT is B transposed ([N][K]). 256 threads = 8 waves; block tile 64x128;
// double-buffered LDS staging via async global->LDS (ASYNCcnt pipeline);
// wave tile 16x64 (4 wmma per k-step).
// MODE 0: bf16 out, 1: gelu->bf16, 2: +residual->f32, 3: f32 out.
// ---------------------------------------------------------------------------
template <int MODE>
__global__ __launch_bounds__(256) void k_gemm(
    const bf16* __restrict__ A, const bf16* __restrict__ BT,
    const float* __restrict__ bias, const float* __restrict__ resid,
    bf16* __restrict__ outB, float* __restrict__ outF,
    int M, int N, int K) {
  __shared__ __align__(32) bf16 sA[2][64 * 32];
  __shared__ __align__(32) bf16 sB[2][128 * 32];
  const int tid = threadIdx.x;
  const int lane = tid & 31, wid = tid >> 5;
  const int mrow = wid >> 1, ncol = wid & 1;
  const int m0 = blockIdx.x * 64, n0 = blockIdx.y * 128;

  v8f acc[4];
#pragma unroll
  for (int j = 0; j < 4; ++j) acc[j] = fzero8();

  const int aRow = (tid * 8) >> 5, aCol = (tid * 8) & 31;
  const int bRow = tid >> 1,       bCol = (tid & 1) * 16;
  const int gmA = m0 + aRow;
  const int gnB = n0 + bRow;
  const bf16* gA = A + (size_t)gmA * K + aCol;   // this thread's A stream
  const bf16* gB = BT + (size_t)gnB * K + bCol;  // this thread's B stream

  auto stage = [&](int k0, int buf) {
    bf16* la = &sA[buf][aRow * 32 + aCol];
    if (gmA < M) {
      lds_copy16(la, gA + k0);
      if (k0 + 32 < K) __builtin_prefetch(gA + k0 + 32, 0, 1);
    } else {
      *(v8bf*)la = bzero8();
    }
    bf16* lb = &sB[buf][bRow * 32 + bCol];
    if (gnB < N) {
      lds_copy16(lb, gB + k0);
      lds_copy16(lb + 8, gB + k0 + 8);
      if (k0 + 32 < K) __builtin_prefetch(gB + k0 + 32, 0, 1);
    } else {
      *(v16bf*)lb = bzero16();
    }
  };

  stage(0, 0);
  wait_async_all();
  __syncthreads();

  int cur = 0;
  for (int k0 = 0; k0 < K; k0 += 32) {
    if (k0 + 32 < K) stage(k0 + 32, cur ^ 1);  // overlap next stage with WMMA
    V16 ua;
    {
      const bf16* pa = &sA[cur][(mrow * 16 + (lane & 15)) * 32 + ((lane < 16) ? 0 : 16)];
      ua.h[0] = *(const v8bf*)pa;
      ua.h[1] = *(const v8bf*)(pa + 8);
    }
#pragma unroll
    for (int j = 0; j < 4; ++j) {
      const bf16* pb = &sB[cur][(ncol * 64 + j * 16 + (lane & 15)) * 32 + (lane >> 4) * 16];
      v16bf vb = *(const v16bf*)pb;
      acc[j] = wmma_bf16(ua.v, vb, acc[j]);
    }
    wait_async_all();
    __syncthreads();
    cur ^= 1;
  }
#pragma unroll
  for (int j = 0; j < 4; ++j) {
#pragma unroll
    for (int r = 0; r < 8; ++r) {
      int m = m0 + mrow * 16 + r + (lane >> 4) * 8;
      int n = n0 + ncol * 64 + j * 16 + (lane & 15);
      if (m < M && n < N) {
        float v = acc[j][r] + bias[n];
        if (MODE == 1) v = gelu(v);
        if (MODE == 2) v += resid[(size_t)m * N + n];
        if (MODE == 0 || MODE == 1) outB[(size_t)m * N + n] = (bf16)v;
        else                        outF[(size_t)m * N + n] = v;
      }
    }
  }
}

// ---------------------------------------------------------------------------
// LayerNorm: fp32 x[rows][768] -> bf16 out (one wave32 per row)
// ---------------------------------------------------------------------------
__global__ __launch_bounds__(256) void k_ln(
    const float* __restrict__ x, const float* __restrict__ g,
    const float* __restrict__ bta, bf16* __restrict__ out, int rows) {
  int row = blockIdx.x * 8 + (threadIdx.x >> 5);
  int lane = threadIdx.x & 31;
  if (row >= rows) return;
  const float* xr = x + (size_t)row * 768;
  float v[24];
  float s = 0.f;
#pragma unroll
  for (int i = 0; i < 24; ++i) { v[i] = xr[lane + i * 32]; s += v[i]; }
#pragma unroll
  for (int o = 16; o > 0; o >>= 1) s += __shfl_xor(s, o, 32);
  float mean = s * (1.f / 768.f);
  float q = 0.f;
#pragma unroll
  for (int i = 0; i < 24; ++i) { float d = v[i] - mean; q += d * d; }
#pragma unroll
  for (int o = 16; o > 0; o >>= 1) q += __shfl_xor(q, o, 32);
  float inv = rsqrtf(q * (1.f / 768.f) + 1e-5f);
  bf16* orow = out + (size_t)row * 768;
#pragma unroll
  for (int i = 0; i < 24; ++i) {
    int c = lane + i * 32;
    orow[c] = (bf16)((v[i] - mean) * inv * g[c] + bta[c]);
  }
}

// ---------------------------------------------------------------------------
// Fused attention per (b, head, 16-query tile), 4 waves.
// Phase1: WMMA scores q.k^T and qer q.er^T into LDS.
// Phase2: skew-trick rel-pos bias + scale + softmax (exact reshape semantics).
// Phase3: WMMA attn @ V from zero-padded V^T.
// ---------------------------------------------------------------------------
__global__ __launch_bounds__(128) void k_attn(
    const bf16* __restrict__ qb, const bf16* __restrict__ kb,
    const bf16* __restrict__ VT, const bf16* __restrict__ er,
    bf16* __restrict__ ob, int T) {
  __shared__ float sS[16][512];   // scores / attn
  __shared__ bf16  sE[17][512];   // qer rows q0..q0+16
  __shared__ float red[16][8];
  const int tid = threadIdx.x;
  const int lane = tid & 31, wid = tid >> 5;
  const int q0 = blockIdx.x * 16;
  const int h = blockIdx.y, b = blockIdx.z;
  const int hb = h * 64;

  // --- Phase 1 ---
  V16 ua[2];
  {
    int qm = q0 + (lane & 15);
    int base = (lane < 16) ? 0 : 16;
#pragma unroll
    for (int ks = 0; ks < 2; ++ks) {
      if (qm < T) {
        const bf16* p = qb + ((size_t)(b * T + qm)) * 768 + hb + ks * 32 + base;
        ua[ks].h[0] = *(const v8bf*)p;
        ua[ks].h[1] = *(const v8bf*)(p + 8);
      } else ua[ks].v = bzero16();
    }
  }
  for (int j = wid; j < 32; j += 4) {
    int n0 = j * 16;
    int tn = n0 + (lane & 15);
    int khalf = (lane >> 4) * 16;
    v8f aS = fzero8(), aE = fzero8();
#pragma unroll
    for (int ks = 0; ks < 2; ++ks) {
      v16bf vk = (tn < T)
          ? *(const v16bf*)(kb + ((size_t)(b * T + tn)) * 768 + hb + ks * 32 + khalf)
          : bzero16();
      aS = wmma_bf16(ua[ks].v, vk, aS);
      v16bf ve = (tn < T)
          ? *(const v16bf*)(er + (size_t)tn * 64 + ks * 32 + khalf)
          : bzero16();
      aE = wmma_bf16(ua[ks].v, ve, aE);
    }
#pragma unroll
    for (int r = 0; r < 8; ++r) {
      int m = r + (lane >> 4) * 8;
      sS[m][tn] = aS[r];
      sE[m][tn] = (bf16)aE[r];
    }
  }
  { // extra qer row (q0+16) needed by the skew wrap of local row 15
    int Qx = q0 + 16;
    const bf16* qp = qb + ((size_t)(b * T + (Qx < T ? Qx : 0))) * 768 + hb;
    for (int c = tid; c < 512; c += 128) {
      float s = 0.f;
      if (Qx < T && c < T) {
        const bf16* ep = er + (size_t)c * 64;
        for (int d = 0; d < 64; ++d) s += (float)qp[d] * (float)ep[d];
      }
      sE[16][c] = (bf16)s;
    }
  }
  __syncthreads();

  // --- Phase 2: srel + scale + softmax ---
  {
    const float scale = 0.125f;  // 1/sqrt(64)
    int r = tid >> 3, c8 = tid & 7;
    int Q = q0 + r;
    float mx = -3.0e38f;
    for (int c = c8; c < 512; c += 8) {
      float vv;
      if (Q < T && c < T) {
        float srel;
        if (c <= Q)            srel = (float)sE[r][c - Q + (T - 1)];
        else if (c == Q + 1)   srel = 0.f;
        else                   srel = (float)sE[r + 1][c - Q - 2];
        vv = (sS[r][c] + srel) * scale;
      } else vv = -3.0e38f;
      sS[r][c] = vv;
      mx = fmaxf(mx, vv);
    }
    red[r][c8] = mx;
    __syncthreads();
    if (c8 == 0) {
      float m2 = red[r][0];
#pragma unroll
      for (int i = 1; i < 8; ++i) m2 = fmaxf(m2, red[r][i]);
      red[r][0] = m2;
    }
    __syncthreads();
    mx = red[r][0];
    float sum = 0.f;
    for (int c = c8; c < 512; c += 8) {
      float e = (Q < T && c < T) ? __expf(sS[r][c] - mx) : 0.f;
      sS[r][c] = e;
      sum += e;
    }
    __syncthreads();
    red[r][c8] = sum;
    __syncthreads();
    if (c8 == 0) {
      float s2 = 0.f;
#pragma unroll
      for (int i = 0; i < 8; ++i) s2 += red[r][i];
      red[r][0] = s2;
    }
    __syncthreads();
    float tot = red[r][0];
    float inv = (tot > 0.f) ? (1.f / tot) : 0.f;
    for (int c = c8; c < 512; c += 8) sS[r][c] *= inv;
  }
  __syncthreads();

  // --- Phase 3: attn @ V (each wave = 16 dh columns) ---
  {
    v8f acc = fzero8();
    int m = lane & 15;
    int base = (lane < 16) ? 0 : 16;
    const bf16* vtb = VT + (((size_t)(b * 12 + h)) * 64 + wid * 16 + (lane & 15)) * 512
                         + (lane >> 4) * 16;
    for (int ks = 0; ks < 16; ++ks) {
      int k0 = ks * 32;
      V16 uA;
#pragma unroll
      for (int jj = 0; jj < 8; ++jj) {
        uA.v[jj]     = (bf16)sS[m][k0 + base + jj];
        uA.v[8 + jj] = (bf16)sS[m][k0 + base + 8 + jj];
      }
      v16bf vb = *(const v16bf*)(vtb + k0);
      acc = wmma_bf16(uA.v, vb, acc);
    }
#pragma unroll
    for (int r = 0; r < 8; ++r) {
      int Q = q0 + r + (lane >> 4) * 8;
      if (Q < T) {
        int dh = wid * 16 + (lane & 15);
        ob[((size_t)(b * T + Q)) * 768 + hb + dh] = (bf16)acc[r];
      }
    }
  }
}

// ---------------------------------------------------------------------------
// Host driver
// ---------------------------------------------------------------------------
extern "C" void kernel_launch(void* const* d_in, const int* in_sizes, int n_in,
                              void* d_out, int out_size, void* d_ws, size_t ws_size,
                              hipStream_t stream) {
  (void)in_sizes; (void)n_in; (void)out_size; (void)ws_size;
  const int B = 16, T = 500, D = 768, FFN = 3072, L = 3;
  const int M = B * T;  // 8000 tokens

  const float* spec   = (const float*)d_in[0];
  const float* cw1    = (const float*)d_in[1];
  const float* cb1    = (const float*)d_in[2];
  const float* cw2    = (const float*)d_in[3];
  const float* cb2    = (const float*)d_in[4];
  const float* cw3    = (const float*)d_in[5];
  const float* cb3    = (const float*)d_in[6];
  const float* cw4    = (const float*)d_in[7];
  const float* cb4    = (const float*)d_in[8];
  const float* proj_w = (const float*)d_in[9];
  const float* proj_b = (const float*)d_in[10];
  const float* ln1_g  = (const float*)d_in[11];
  const float* ln1_b  = (const float*)d_in[12];
  const float* wq     = (const float*)d_in[13];
  const float* bq     = (const float*)d_in[14];
  const float* wk     = (const float*)d_in[15];
  const float* bk     = (const float*)d_in[16];
  const float* wv     = (const float*)d_in[17];
  const float* bv     = (const float*)d_in[18];
  const float* wo     = (const float*)d_in[19];
  const float* bo     = (const float*)d_in[20];
  const float* er     = (const float*)d_in[21];
  const float* ln2_g  = (const float*)d_in[22];
  const float* ln2_b  = (const float*)d_in[23];
  const float* w1     = (const float*)d_in[24];
  const float* b1     = (const float*)d_in[25];
  const float* w2     = (const float*)d_in[26];
  const float* b2     = (const float*)d_in[27];
  const float* depw   = (const float*)d_in[28];
  const float* depb   = (const float*)d_in[29];

  size_t off = 0;
  auto alloc = [&](size_t bytes) -> void* {
    off = (off + 255) & ~(size_t)255;
    void* p = (char*)d_ws + off;
    off += bytes;
    return p;
  };
  const size_t DD = (size_t)D * D;                 // 589824
  bf16* projWT = (bf16*)alloc(DD * 2);
  bf16* wqT    = (bf16*)alloc(3 * DD * 2);
  bf16* wkT    = (bf16*)alloc(3 * DD * 2);
  bf16* wvT    = (bf16*)alloc(3 * DD * 2);
  bf16* woT    = (bf16*)alloc(3 * DD * 2);
  bf16* w1T    = (bf16*)alloc((size_t)3 * D * FFN * 2);
  bf16* w2T    = (bf16*)alloc((size_t)3 * FFN * D * 2);
  bf16* depWT  = (bf16*)alloc((size_t)D * 192 * 2);
  bf16* er_bf  = (bf16*)alloc((size_t)3 * T * 64 * 2);
  bf16* pack2  = (bf16*)alloc((size_t)25 * 1 * 4 * 512 * 2);
  bf16* pack3  = (bf16*)alloc((size_t)25 * 2 * 4 * 512 * 2);
  bf16* pack4  = (bf16*)alloc((size_t)25 * 2 * 4 * 512 * 2);
  bf16* X1     = (bf16*)alloc((size_t)B * 96 * T * 32 * 2);
  bf16* X2     = (bf16*)alloc((size_t)B * 48 * T * 64 * 2);
  bf16* X3     = (bf16*)alloc((size_t)B * 24 * T * 64 * 2);
  bf16* X4     = (bf16*)alloc((size_t)B * 12 * T * 64 * 2);
  bf16* tok_bf = (bf16*)alloc((size_t)M * D * 2);
  float* x     = (float*)alloc((size_t)M * D * 4);
  bf16* xn_bf  = (bf16*)alloc((size_t)M * D * 2);
  bf16* q_bf   = (bf16*)alloc((size_t)M * D * 2);
  bf16* k_bf   = (bf16*)alloc((size_t)M * D * 2);
  bf16* v_bf   = (bf16*)alloc((size_t)M * D * 2);
  bf16* VT     = (bf16*)alloc((size_t)B * 12 * 64 * 512 * 2);
  bf16* o_bf   = (bf16*)alloc((size_t)M * D * 2);
  bf16* h_bf   = (bf16*)alloc((size_t)M * FFN * 2);

  auto g1 = [](size_t n) { return dim3((unsigned)((n + 255) / 256)); };

  // ---- weight conversions (per launch; deterministic) ----
  k_wt<<<g1(DD), 256, 0, stream>>>(proj_w, projWT, D, D);
  for (int i = 0; i < L; ++i) {
    k_wt<<<g1(DD), 256, 0, stream>>>(wq + i * DD, wqT + i * DD, D, D);
    k_wt<<<g1(DD), 256, 0, stream>>>(wk + i * DD, wkT + i * DD, D, D);
    k_wt<<<g1(DD), 256, 0, stream>>>(wv + i * DD, wvT + i * DD, D, D);
    k_wt<<<g1(DD), 256, 0, stream>>>(wo + i * DD, woT + i * DD, D, D);
    k_wt<<<g1((size_t)D * FFN), 256, 0, stream>>>(w1 + (size_t)i * D * FFN,
                                                  w1T + (size_t)i * FFN * D, D, FFN);
    k_wt<<<g1((size_t)FFN * D), 256, 0, stream>>>(w2 + (size_t)i * FFN * D,
                                                  w2T + (size_t)i * D * FFN, FFN, D);
  }
  k_wt<<<g1((size_t)D * 192), 256, 0, stream>>>(depw, depWT, D, 192);
  k_cvt<<<g1((size_t)3 * T * 64), 256, 0, stream>>>(er, er_bf, (size_t)3 * T * 64);
  k_pack_conv<<<g1(25 * 1 * 4 * 512), 256, 0, stream>>>(cw2, pack2, 32);
  k_pack_conv<<<g1(25 * 2 * 4 * 512), 256, 0, stream>>>(cw3, pack3, 64);
  k_pack_conv<<<g1(25 * 2 * 4 * 512), 256, 0, stream>>>(cw4, pack4, 64);

  // ---- CNN front-end ----
  k_conv1<<<dim3((T + 127) / 128, 96, B), 128, 0, stream>>>(spec, cw1, cb1, X1, T);
  k_conv_wmma<32, 96><<<dim3(32, 48, B), 128, 0, stream>>>(X1, pack2, cb2, X2, T);
  k_conv_wmma<64, 48><<<dim3(32, 24, B), 128, 0, stream>>>(X2, pack3, cb3, X3, T);
  k_conv_wmma<64, 24><<<dim3(32, 12, B), 128, 0, stream>>>(X3, pack4, cb4, X4, T);
  k_tok<<<g1((size_t)M * D), 256, 0, stream>>>(X4, tok_bf, T);

  // ---- projection into residual stream (fp32) ----
  k_gemm<3><<<dim3(M / 64, (D + 127) / 128), 256, 0, stream>>>(
      tok_bf, projWT, proj_b, nullptr, nullptr, x, M, D, D);

  // ---- transformer layers ----
  for (int i = 0; i < L; ++i) {
    k_ln<<<dim3(M / 8), 256, 0, stream>>>(x, ln1_g + i * D, ln1_b + i * D, xn_bf, M);
    k_gemm<0><<<dim3(M / 64, (D + 127) / 128), 256, 0, stream>>>(
        xn_bf, wqT + i * DD, bq + i * D, nullptr, q_bf, nullptr, M, D, D);
    k_gemm<0><<<dim3(M / 64, (D + 127) / 128), 256, 0, stream>>>(
        xn_bf, wkT + i * DD, bk + i * D, nullptr, k_bf, nullptr, M, D, D);
    k_gemm<0><<<dim3(M / 64, (D + 127) / 128), 256, 0, stream>>>(
        xn_bf, wvT + i * DD, bv + i * D, nullptr, v_bf, nullptr, M, D, D);
    k_vt<<<g1((size_t)B * 12 * 64 * 512), 256, 0, stream>>>(v_bf, VT, T);
    k_attn<<<dim3(32, 12, B), 128, 0, stream>>>(
        q_bf, k_bf, VT, er_bf + (size_t)i * T * 64, o_bf, T);
    k_gemm<2><<<dim3(M / 64, (D + 127) / 128), 256, 0, stream>>>(
        o_bf, woT + i * DD, bo + i * D, x, nullptr, x, M, D, D);
    k_ln<<<dim3(M / 8), 256, 0, stream>>>(x, ln2_g + i * D, ln2_b + i * D, xn_bf, M);
    k_gemm<1><<<dim3(M / 64, (FFN + 127) / 128), 256, 0, stream>>>(
        xn_bf, w1T + (size_t)i * FFN * D, b1 + i * FFN, nullptr, h_bf, nullptr, M, FFN, D);
    k_gemm<2><<<dim3(M / 64, (D + 127) / 128), 256, 0, stream>>>(
        h_bf, w2T + (size_t)i * D * FFN, b2 + i * D, x, nullptr, x, M, D, FFN);
  }

  // ---- deprojection -> fp32 output ----
  k_cvt<<<g1((size_t)M * D), 256, 0, stream>>>(x, xn_bf, (size_t)M * D);
  k_gemm<3><<<dim3(M / 64, (192 + 127) / 128), 256, 0, stream>>>(
      xn_bf, depWT, depb, nullptr, nullptr, (float*)d_out, M, 192, D);
}
